// RC3D_25151328485468
// MI455X (gfx1250) — matrix operand order repaired and checked by reading, not compiled, not run on hardware
//
#include <hip/hip_runtime.h>

typedef unsigned short u16;
typedef __attribute__((ext_vector_type(16))) __bf16 v16bf;
typedef __attribute__((ext_vector_type(8)))  float  v8f;

#define CEIL(a,b) (((a)+(b)-1)/(b))
#define NEG_INF (-__builtin_inff())

// ---------------- helpers ----------------
__device__ __forceinline__ u16 f2bf(float f) {
  unsigned u = __float_as_uint(f);
  unsigned r = u + 0x7FFFu + ((u >> 16) & 1u);   // RNE
  return (u16)(r >> 16);
}
__device__ __forceinline__ unsigned ford(float f) { // monotone float->uint
  unsigned u = __float_as_uint(f);
  return (u & 0x80000000u) ? ~u : (u | 0x80000000u);
}

// ---------------- generic converts ----------------
__global__ void rc3d_cvt(const float* __restrict__ in, u16* __restrict__ out, int n) {
  int i = blockIdx.x * blockDim.x + threadIdx.x;
  if (i < n) out[i] = f2bf(in[i]);
}
// zero-padded rows: out is [rowsPad, cols]
__global__ void rc3d_cvt_pad(const float* __restrict__ in, u16* __restrict__ out,
                             int rows, int cols, int rowsPad) {
  int i = blockIdx.x * blockDim.x + threadIdx.x;
  if (i >= rowsPad * cols) return;
  int r = i / cols;
  out[i] = (r < rows) ? f2bf(in[i]) : (u16)0;
}
// out[c*rows + r] = in[r*cols + c]   (transpose + convert, exact)
__global__ void rc3d_cvt_tr(const float* __restrict__ in, u16* __restrict__ out,
                            int rows, int cols) {
  int i = blockIdx.x * blockDim.x + threadIdx.x;
  if (i >= rows * cols) return;
  int r = i / cols, c = i % cols;
  out[c * rows + r] = f2bf(in[i]);
}
// transpose + convert with zero padding: out is [colsPad, ldo]; out[co*ldo+ro]=in[ro*cols+co]
__global__ void rc3d_cvt_tr_pad(const float* __restrict__ in, u16* __restrict__ out,
                                int rows, int cols, int ldo, int colsPad) {
  int i = blockIdx.x * blockDim.x + threadIdx.x;
  if (i >= colsPad * ldo) return;
  int co = i / ldo, ro = i % ldo;
  out[i] = (co < cols && ro < rows) ? f2bf(in[ro * cols + co]) : (u16)0;
}

// ---------------- WMMA GEMM:  C[M,N] = A[M,K] * B[N,K]^T  ----------------
// Requirements (guaranteed by staging): K % 32 == 0; A has >= ceil(M/32)*32 rows,
// B has >= ceil(N/32)*32 rows; K-padding columns are zero; lda/ldb % 8 == 0.
// One wave computes a 32x32 C tile: 8 x b128 loads + 4 x WMMA per K step.
__global__ void rc3d_gemm(const u16* __restrict__ A, int lda,
                          const u16* __restrict__ B, int ldb,
                          float* __restrict__ C, int ldc,
                          const float* __restrict__ bias, int biasMode, int doRelu,
                          int M, int N, int K)
{
  const int lane = threadIdx.x & 31;
  const int r15 = lane & 15, kh = lane >> 4;
  const int m0 = blockIdx.x * 32, n0 = blockIdx.y * 32;
  const u16* pa0 = A + (size_t)(m0 + r15) * lda;
  const u16* pa1 = pa0 + (size_t)16 * lda;
  const u16* pb0 = B + (size_t)(n0 + r15) * ldb;
  const u16* pb1 = pb0 + (size_t)16 * ldb;
  v8f acc00 = {}, acc01 = {}, acc10 = {}, acc11 = {};
  for (int k0 = 0; k0 < K; k0 += 32) {
    __builtin_prefetch(pa0 + k0 + 512, 0, 1);
    __builtin_prefetch(pb0 + k0 + 512, 0, 1);
    union { uint4 q[2]; v16bf v; } a0, a1, b0, b1;
    // A fragment (16x32): vi 0..3 -> elems [k0+kh*8, +8); vi 4..7 -> [k0+16+kh*8, +8)
    a0.q[0] = *(const uint4*)(pa0 + k0 + (kh << 3));
    a0.q[1] = *(const uint4*)(pa0 + k0 + 16 + (kh << 3));
    a1.q[0] = *(const uint4*)(pa1 + k0 + (kh << 3));
    a1.q[1] = *(const uint4*)(pa1 + k0 + 16 + (kh << 3));
    // B fragment (32x16): vi 0..3 -> elems [k0+kh*8, +8); vi 4..7 -> [k0+16+kh*8, +8)
    b0.q[0] = *(const uint4*)(pb0 + k0 + (kh << 3));
    b0.q[1] = *(const uint4*)(pb0 + k0 + 16 + (kh << 3));
    b1.q[0] = *(const uint4*)(pb1 + k0 + (kh << 3));
    b1.q[1] = *(const uint4*)(pb1 + k0 + 16 + (kh << 3));
    acc00 = __builtin_amdgcn_wmma_f32_16x16x32_bf16(false, a0.v, false, b0.v, (short)0, acc00, false, false);
    acc01 = __builtin_amdgcn_wmma_f32_16x16x32_bf16(false, a0.v, false, b1.v, (short)0, acc01, false, false);
    acc10 = __builtin_amdgcn_wmma_f32_16x16x32_bf16(false, a1.v, false, b0.v, (short)0, acc10, false, false);
    acc11 = __builtin_amdgcn_wmma_f32_16x16x32_bf16(false, a1.v, false, b1.v, (short)0, acc11, false, false);
  }
  auto st = [&](v8f acc, int mb, int nb) {
    int col = nb + r15;
    if (col < N) {
#pragma unroll
      for (int r = 0; r < 8; ++r) {
        int m = mb + r + (kh << 3);      // C/D layout: vgpr r -> M=r (+8 for lanes 16-31)
        if (m < M) {
          float v = acc[r];
          if (biasMode == 1) v += bias[m];
          else if (biasMode == 2) v += bias[col];
          if (doRelu) v = fmaxf(v, 0.0f);
          C[(size_t)m * ldc + col] = v;
        }
      }
    }
  };
  st(acc00, m0, n0);
  st(acc01, m0, n0 + 16);
  st(acc10, m0 + 16, n0);
  st(acc11, m0 + 16, n0 + 16);
}

// ---------------- stage kernels ----------------
// im2col^T for rpn_conv: out[l, i*3+t] = x[i, l+t-1] (zero pad)
__global__ void rc3d_im2colT(const float* __restrict__ x, u16* __restrict__ out) {
  int i = blockIdx.x * blockDim.x + threadIdx.x;
  if (i >= 1024 * 768) return;
  int l = i / 768, r = i % 768, ch = r / 3, t = r % 3;
  int pos = l + t - 1;
  float v = (pos >= 0 && pos < 1024) ? x[ch * 1024 + pos] : 0.0f;
  out[i] = f2bf(v);
}

// anchor decode + pairwise softmax over (L,K,2)
__global__ void rc3d_scores(const float* __restrict__ cls, const float* __restrict__ seg,
                            float* sc, float* cxv, float* hlv, float* sv, float* ev) {
  int i = blockIdx.x * blockDim.x + threadIdx.x;
  if (i >= 10240) return;
  const float ah[10] = {8.f,16.f,20.f,24.f,32.f,36.f,40.f,48.f,56.f,64.f};
  int l = i / 10, k = i % 10;
  float c0 = cls[(2 * k) * 1024 + l], c1 = cls[(2 * k + 1) * 1024 + l];
  float mx = fmaxf(c0, c1);
  float e0 = expf(c0 - mx), e1 = expf(c1 - mx);
  sc[i] = e1 / (e0 + e1);
  float o0 = seg[(2 * k) * 1024 + l], o1 = seg[(2 * k + 1) * 1024 + l];
  float acx = (float)l * 8.0f + 4.0f, half = ah[k];
  float cx = half * o0 + acx;
  float hl = expf(o1) * half;
  cx = fminf(fmaxf(cx, 0.0f), 8191.0f);
  hl = fminf(fmaxf(hl, 1.0f), 8192.0f);
  cxv[i] = cx; hlv[i] = hl;
  sv[i] = fminf(fmaxf(cx - hl, 0.0f), 8191.0f);
  ev[i] = fminf(fmaxf(cx + hl, 0.0f), 8191.0f);
}

// exact rank-select top-1024 (unique composite keys => deterministic sorted order)
__global__ void rc3d_topk(const float* __restrict__ sc, int* __restrict__ tidx) {
  int i = blockIdx.x * blockDim.x + threadIdx.x;
  if (i >= 10240) return;
  unsigned long long ki = ((unsigned long long)ford(sc[i]) << 32) | (0xFFFFFFFFu - (unsigned)i);
  int rank = 0;
  for (int j = 0; j < 10240; ++j) {
    unsigned long long kj = ((unsigned long long)ford(sc[j]) << 32) | (0xFFFFFFFFu - (unsigned)j);
    rank += (kj > ki);
  }
  if (rank < 1024) tidx[rank] = i;
}

// serial 1D NMS over the sorted top-1024, then select POST_NMS=300
__global__ void rc3d_nms(const int* __restrict__ tidx,
                         const float* __restrict__ sv, const float* __restrict__ ev,
                         const float* __restrict__ cxv, const float* __restrict__ hlv,
                         int* __restrict__ pidx, float* __restrict__ bbc, float* __restrict__ bbh) {
  __shared__ float ss[1024], ee[1024], ll[1024];
  __shared__ int keep[1024];
  int j = threadIdx.x;
  int p = tidx[j];
  ss[j] = sv[p]; ee[j] = ev[p]; ll[j] = ev[p] - sv[p]; keep[j] = 1;
  __syncthreads();
  for (int i = 0; i < 1024; ++i) {
    if (keep[i] && j > i) {
      float inter = fmaxf(fminf(ee[i], ee[j]) - fmaxf(ss[i], ss[j]), 0.0f);
      float iou = inter / fmaxf(ll[i] + ll[j] - inter, 1e-6f);
      if (iou > 0.7f) keep[j] = 0;
    }
    __syncthreads();
  }
  if (j == 0) {
    int cnt = 0;
    for (int t = 0; t < 1024 && cnt < 300; ++t)
      if (keep[t]) { int q = tidx[t]; pidx[cnt] = q; bbc[cnt] = cxv[q]; bbh[cnt] = hlv[q]; ++cnt; }
    for (int t = 0; t < 1024 && cnt < 300; ++t)
      if (!keep[t]) { int q = tidx[t]; pidx[cnt] = q; bbc[cnt] = cxv[q]; bbh[cnt] = hlv[q]; ++cnt; }
  }
}

__global__ void rc3d_gather_props(const float* __restrict__ cls, const float* __restrict__ seg,
                                  const int* __restrict__ pidx, float* __restrict__ out) {
  int n = blockIdx.x * blockDim.x + threadIdx.x;
  if (n >= 300) return;
  int p = pidx[n], l = p / 10, k = p % 10;
  out[n * 2 + 0]       = cls[(2 * k) * 1024 + l];
  out[n * 2 + 1]       = cls[(2 * k + 1) * 1024 + l];
  out[600 + n * 2 + 0] = seg[(2 * k) * 1024 + l];
  out[600 + n * 2 + 1] = seg[(2 * k + 1) * 1024 + l];
}

__global__ void rc3d_sidx(const float* __restrict__ bbc, const float* __restrict__ bbh,
                          int* __restrict__ sidx, int* __restrict__ eidx) {
  int n = blockIdx.x * blockDim.x + threadIdx.x;
  if (n >= 300) return;
  int s = (int)floorf((bbc[n] - bbh[n]) * 0.125f);
  int e = (int)ceilf((bbc[n] + bbh[n]) * 0.125f);
  s = min(max(s, 0), 1023);
  e = min(max(e, 0), 1023);
  if (e < s) e = s;
  sidx[n] = s; eidx[n] = e;
}

// ROI adaptive maxpool (two pow2 windows == sparse-table max). out Bt[(n*7+j), c]
__global__ void rc3d_roipool(const float* __restrict__ feat, const int* __restrict__ sidx,
                             const int* __restrict__ eidx, u16* __restrict__ spp) {
  int b = blockIdx.x, n = b / 7, j = b % 7;
  int s = sidx[n], e = eidx[n];
  int len = e - s + 1;
  int bs = s + (j * len) / 7;
  int be = s + ((j + 1) * len + 6) / 7;
  int wlen = max(be - bs, 1);
  int kp = (int)floorf(log2f((float)wlen) + 1e-4f);
  int win = 1 << kp;
  int a = bs, b2 = be - win; if (b2 < 0) b2 = 0;
  for (int c = threadIdx.x; c < 1024; c += blockDim.x) {
    const float* fr = feat + c * 1024;
    float mv = NEG_INF;
    for (int t = 0; t < win; ++t) {
      int p1 = a + t;  if (p1 < 1024) mv = fmaxf(mv, fr[p1]);
      int p2 = b2 + t; if (p2 < 1024) mv = fmaxf(mv, fr[p2]);
    }
    spp[(n * 7 + j) * 1024 + c] = f2bf(mv);
  }
}

// x2flat[n, o*7+j] = Y[o, n*7+j]
__global__ void rc3d_x2flat(const float* __restrict__ Y, u16* __restrict__ out) {
  int i = blockIdx.x * blockDim.x + threadIdx.x;
  if (i >= 300 * 1792) return;
  int n = i / 1792, r = i % 1792, o = r / 7, j = r % 7;
  out[i] = f2bf(Y[o * 2100 + n * 7 + j]);
}

// relation position embedding (300x300x64) to bf16
__global__ void rc3d_posemb(const float* __restrict__ bbc, const float* __restrict__ bbh,
                            u16* __restrict__ embb) {
  int i = blockIdx.x * blockDim.x + threadIdx.x;
  if (i >= 90000) return;
  int n = i / 300, m = i % 300;
  float lnn = fmaxf(2.0f * bbh[n], 1.0f);
  float lm  = fmaxf(2.0f * bbh[m], 1.0f);
  float dd = logf(fmaxf(fabsf(bbc[n] - bbc[m]), 1e-3f) / lnn);
  float rr = logf(lm / lnn);
  u16* o = embb + (size_t)i * 64;
#pragma unroll
  for (int f = 0; f < 16; ++f) {
    float dinv = 100.0f / powf(1000.0f, (float)f * (1.0f / 16.0f));
    float vd = dd * dinv, vr = rr * dinv;
    o[f]      = f2bf(sinf(vd)); o[16 + f] = f2bf(cosf(vd));
    o[32 + f] = f2bf(sinf(vr)); o[48 + f] = f2bf(cosf(vr));
  }
}

// softmax over m for relation attention: sm[h, n, m] with K padded to 320 (zeros)
__global__ void rc3d_rel_softmax(const float* __restrict__ affw, const float* __restrict__ qb,
                                 const float* __restrict__ kb, u16* __restrict__ smb) {
  __shared__ float wv_s[512];
  __shared__ float red[512];
  int h = blockIdx.x / 300, n = blockIdx.x % 300;
  int m = threadIdx.x;
  float wv = NEG_INF;
  if (m < 300) {
    float aw = affw[(n * 300 + m) * 16 + h];           // relu'd by GEMM epilogue
    const float* qq = qb + n * 256 + h * 16;
    const float* kk = kb + m * 256 + h * 16;
    float qk = 0.0f;
#pragma unroll
    for (int d = 0; d < 16; ++d) qk += qq[d] * kk[d];
    wv = logf(fmaxf(aw, 1e-6f)) + qk * 0.25f;
  }
  wv_s[m] = wv; red[m] = wv; __syncthreads();
  for (int s = 256; s > 0; s >>= 1) { if (m < s) red[m] = fmaxf(red[m], red[m + s]); __syncthreads(); }
  float mx = red[0]; __syncthreads();
  float e = (m < 300) ? expf(wv_s[m] - mx) : 0.0f;
  red[m] = e; __syncthreads();
  for (int s = 256; s > 0; s >>= 1) { if (m < s) red[m] += red[m + s]; __syncthreads(); }
  float inv = 1.0f / red[0];
  if (m < 300)      smb[(size_t)h * 102400 + n * 320 + m] = f2bf(e * inv);
  else if (m < 320) smb[(size_t)h * 102400 + n * 320 + m] = (u16)0;   // K padding
}

__global__ void rc3d_addrelu_x2(const float* __restrict__ x2pre, const float* __restrict__ relout,
                                const float* __restrict__ relb, float* __restrict__ x2r,
                                u16* __restrict__ x2rb) {
  int i = blockIdx.x * blockDim.x + threadIdx.x;
  if (i >= 300 * 256) return;
  float v = fmaxf(x2pre[i] + relout[i] + relb[i % 256], 0.0f);
  x2r[i] = v; x2rb[i] = f2bf(v);
}

__global__ void rc3d_prob_softmax(const float* __restrict__ ocs, float* __restrict__ prob) {
  int n = blockIdx.x * blockDim.x + threadIdx.x;
  if (n >= 300) return;
  const float* x = ocs + n * 21;
  float mx = x[0];
  for (int c = 1; c < 21; ++c) mx = fmaxf(mx, x[c]);
  float s = 0.0f;
  for (int c = 0; c < 21; ++c) s += expf(x[c] - mx);
  float inv = 1.0f / s;
  for (int c = 0; c < 20; ++c) prob[n * 20 + c] = expf(x[c + 1] - mx) * inv;
}

// per-class stable ascending sort (bitonic over 512)
__global__ void rc3d_class_sort(const float* __restrict__ prob, int* __restrict__ sidx,
                                float* __restrict__ ssc) {
  __shared__ unsigned long long keys[512];
  int c = blockIdx.x, t = threadIdx.x;
  keys[t] = (t < 300)
    ? (((unsigned long long)ford(prob[t * 20 + c]) << 32) | (unsigned)t)
    : 0xFFFFFFFFFFFFFFFFull;
  __syncthreads();
  for (int ksz = 2; ksz <= 512; ksz <<= 1)
    for (int j = ksz >> 1; j > 0; j >>= 1) {
      int ixj = t ^ j;
      if (ixj > t) {
        bool up = ((t & ksz) == 0);
        unsigned long long a = keys[t], b = keys[ixj];
        if ((a > b) == up) { keys[t] = b; keys[ixj] = a; }
      }
      __syncthreads();
    }
  if (t < 300) {
    int idx = (int)(keys[t] & 0xFFFFFFFFull);
    sidx[t * 20 + c] = idx;
    ssc[t * 20 + c] = prob[idx * 20 + c];
  }
}

__global__ void rc3d_refined(const float* __restrict__ ooff, const float* __restrict__ bbc,
                             const float* __restrict__ bbh, float* __restrict__ refined) {
  int i = blockIdx.x * blockDim.x + threadIdx.x;
  if (i >= 6000) return;
  int n = i / 20, c = i % 20;
  float o0 = ooff[n * 40 + c * 2], o1 = ooff[n * 40 + c * 2 + 1];
  float rc = bbh[n] * o0 + bbc[n];
  float rl = expf(o1) * bbh[n];
  refined[i * 2 + 0] = fminf(fmaxf(rc - rl * 0.5f, 0.0f), 8191.0f);
  refined[i * 2 + 1] = fminf(fmaxf(rc + rl * 0.5f, 0.0f), 8191.0f);
}

__global__ void rc3d_sorted_bbox(const int* __restrict__ sidx, const float* __restrict__ refined,
                                 float* __restrict__ sbb) {
  int i = blockIdx.x * blockDim.x + threadIdx.x;
  if (i >= 6000) return;
  int r = i / 20, c = i % 20, m = sidx[r * 20 + c];
  sbb[i * 2 + 0] = refined[(m * 20 + c) * 2 + 0];
  sbb[i * 2 + 1] = refined[(m * 20 + c) * 2 + 1];
}

__global__ void rc3d_rank_emb(u16* __restrict__ out) {
  int i = blockIdx.x * blockDim.x + threadIdx.x;
  if (i >= 300 * 256) return;
  int n = i / 256, f = i % 256;
  float v = (f < 128)
    ? sinf((float)n / powf(1000.0f, (float)f * (1.0f / 128.0f)))
    : cosf((float)n / powf(1000.0f, (float)(f - 128) * (1.0f / 128.0f)));
  out[i] = f2bf(v);
}

__global__ void rc3d_sroi_nmsemb(const int* __restrict__ sidx, const float* __restrict__ roiemb,
                                 const float* __restrict__ nmsrank,
                                 float* __restrict__ sroi, u16* __restrict__ sroib,
                                 float* __restrict__ nemb, u16* __restrict__ nembb) {
  int i = blockIdx.x * blockDim.x + threadIdx.x;
  if (i >= 6000 * 128) return;
  int rc = i / 128, d = i % 128, r = rc / 20, c = rc % 20;
  int m = sidx[r * 20 + c];
  float v = roiemb[m * 128 + d];
  sroi[i] = v; sroib[i] = f2bf(v);
  float ne = v + nmsrank[r * 128 + d];
  nemb[i] = ne; nembb[i] = f2bf(ne);
}

// fused NMS-relation attention: per (rank n, class c, head h) online softmax over m
__global__ void rc3d_nms_attn(const float* __restrict__ sbb, const float* __restrict__ q2,
                              const float* __restrict__ k2, const float* __restrict__ vp,
                              const float* __restrict__ npw, const float* __restrict__ npb,
                              float* __restrict__ att2) {
  int i = blockIdx.x * blockDim.x + threadIdx.x;
  if (i >= 96000) return;
  int n = i / 320, rem = i % 320, c = rem / 16, h = rem % 16;
  float s0 = sbb[(n * 20 + c) * 2], e0 = sbb[(n * 20 + c) * 2 + 1];
  float cxn = 0.5f * (s0 + e0), ll = fmaxf(e0 - s0, 1.0f);
  float Wh[64], dinv[16], q8[8];
  for (int f = 0; f < 64; ++f) Wh[f] = npw[h * 64 + f];
  for (int f = 0; f < 16; ++f) dinv[f] = 100.0f / powf(1000.0f, (float)f * (1.0f / 16.0f));
  for (int d = 0; d < 8; ++d) q8[d] = q2[(n * 20 + c) * 128 + h * 8 + d];
  float bh = npb[h];
  float mx = NEG_INF, ssum = 0.0f, acc[8] = {0, 0, 0, 0, 0, 0, 0, 0};
  for (int m = 0; m < 300; ++m) {
    float s1 = sbb[(m * 20 + c) * 2], e1 = sbb[(m * 20 + c) * 2 + 1];
    float cxm = 0.5f * (s1 + e1), lm = fmaxf(e1 - s1, 1.0f);
    float dd = logf(fmaxf(fabsf(cxn - cxm), 1e-3f) / ll);
    float rr = logf(lm / ll);
    float aw = bh;
#pragma unroll
    for (int f = 0; f < 16; ++f) {
      float vd = dd * dinv[f], vr = rr * dinv[f];
      aw += sinf(vd) * Wh[f] + cosf(vd) * Wh[16 + f] + sinf(vr) * Wh[32 + f] + cosf(vr) * Wh[48 + f];
    }
    float wv = logf(fmaxf(fmaxf(aw, 0.0f), 1e-6f));
    const float* kk = k2 + (m * 20 + c) * 128 + h * 8;
    float qk = 0.0f;
#pragma unroll
    for (int d = 0; d < 8; ++d) qk += q8[d] * kk[d];
    wv += qk * 0.35355339059f;
    const float* vv = vp + (m * 20 + c) * 128 + h * 8;
    if (wv > mx) {
      float scl = expf(mx - wv);
      ssum *= scl;
#pragma unroll
      for (int o = 0; o < 8; ++o) acc[o] *= scl;
      mx = wv; ssum += 1.0f;
#pragma unroll
      for (int o = 0; o < 8; ++o) acc[o] += vv[o];
    } else {
      float ee = expf(wv - mx);
      ssum += ee;
#pragma unroll
      for (int o = 0; o < 8; ++o) acc[o] += ee * vv[o];
    }
  }
  float inv = 1.0f / ssum;
#pragma unroll
  for (int o = 0; o < 8; ++o) att2[(n * 20 + c) * 128 + h * 8 + o] = acc[o] * inv;
}

__global__ void rc3d_final(const float* __restrict__ nemb, const float* __restrict__ att2,
                           const float* __restrict__ nob, const float* __restrict__ fc3w,
                           const float* __restrict__ fc3b, const float* __restrict__ ssc,
                           float* __restrict__ outns) {
  int i = blockIdx.x * blockDim.x + threadIdx.x;
  if (i >= 6000) return;
  const float* a = nemb + i * 128;
  const float* b = att2 + i * 128;
  float acc = 0.0f;
  for (int d = 0; d < 128; ++d) {
    float v = fmaxf(a[d] + b[d] + nob[d], 0.0f);
    acc += v * fc3w[d];
  }
  float logit = acc + fc3b[0];
  outns[i] = (1.0f / (1.0f + expf(-logit))) * ssc[i];
}

// ---------------- host glue ----------------
static void gemm(const u16* A, int lda, const u16* B, int ldb, float* C, int ldc,
                 const float* bias, int bm, int relu, int M, int N, int K, hipStream_t s) {
  dim3 g(CEIL(M, 32), CEIL(N, 32));
  rc3d_gemm<<<g, 32, 0, s>>>(A, lda, B, ldb, C, ldc, bias, bm, relu, M, N, K);
}

extern "C" void kernel_launch(void* const* d_in, const int* in_sizes, int n_in,
                              void* d_out, int out_size, void* d_ws, size_t ws_size,
                              hipStream_t stream) {
  const float* feature   = (const float*)d_in[0];
  const float* conv1_w   = (const float*)d_in[1];  const float* conv1_b   = (const float*)d_in[2];
  const float* rpnc_w    = (const float*)d_in[3];  const float* rpnc_b    = (const float*)d_in[4];
  const float* rcls_w    = (const float*)d_in[5];  const float* rcls_b    = (const float*)d_in[6];
  const float* rseg_w    = (const float*)d_in[7];  const float* rseg_b    = (const float*)d_in[8];
  const float* conv2_w   = (const float*)d_in[9];  const float* conv2_b   = (const float*)d_in[10];
  const float* fc1_w     = (const float*)d_in[11]; const float* fc1_b     = (const float*)d_in[12];
  const float* cls_w     = (const float*)d_in[13]; const float* cls_b     = (const float*)d_in[14];
  const float* bbox_w    = (const float*)d_in[15]; const float* bbox_b    = (const float*)d_in[16];
  const float* fc1n_w    = (const float*)d_in[17]; const float* fc1n_b    = (const float*)d_in[18];
  const float* fc2n_w    = (const float*)d_in[19]; const float* fc2n_b    = (const float*)d_in[20];
  const float* fc3n_w    = (const float*)d_in[21]; const float* fc3n_b    = (const float*)d_in[22];
  const float* relpos_w  = (const float*)d_in[23]; const float* relpos_b  = (const float*)d_in[24];
  const float* relq_w    = (const float*)d_in[25]; const float* relq_b    = (const float*)d_in[26];
  const float* relk_w    = (const float*)d_in[27]; const float* relk_b    = (const float*)d_in[28];
  const float* relout_w  = (const float*)d_in[29]; const float* relout_b  = (const float*)d_in[30];
  const float* nrelpos_w = (const float*)d_in[31]; const float* nrelpos_b = (const float*)d_in[32];
  const float* nrelq_w   = (const float*)d_in[33]; const float* nrelq_b   = (const float*)d_in[34];
  const float* nrelk_w   = (const float*)d_in[35]; const float* nrelk_b   = (const float*)d_in[36];
  const float* nrelout_w = (const float*)d_in[37]; const float* nrelout_b = (const float*)d_in[38];
  float* out = (float*)d_out;

  char* w = (char*)d_ws;
  size_t off = 0;
  auto alloc = [&](size_t bytes) -> void* {
    void* p = w + off; off += (bytes + 255) & ~(size_t)255; return p;
  };

  // bf16 staging (rows padded to multiples of 32 where needed; K pads zero-filled)
  u16* featTb  = (u16*)alloc(1024 * 1024 * 2);
  u16* w1b     = (u16*)alloc(256 * 1024 * 2);
  u16* wrb     = (u16*)alloc(256 * 768 * 2);
  u16* clswb   = (u16*)alloc(32 * 256 * 2);          // 20 -> 32 rows
  u16* segwb   = (u16*)alloc(32 * 256 * 2);
  u16* w2b     = (u16*)alloc(256 * 1024 * 2);
  u16* fc1wb   = (u16*)alloc(256 * 1792 * 2);
  u16* clsw21b = (u16*)alloc(32 * 256 * 2);          // 21 -> 32 rows
  u16* bboxwb  = (u16*)alloc(64 * 256 * 2);          // 40 -> 64 rows
  u16* fc1nwb  = (u16*)alloc(128 * 256 * 2);
  u16* fc2nwb  = (u16*)alloc(128 * 256 * 2);
  u16* rposwb  = (u16*)alloc(32 * 64 * 2);           // 16 -> 32 rows
  u16* rqwb    = (u16*)alloc(256 * 256 * 2);
  u16* rkwb    = (u16*)alloc(256 * 256 * 2);
  u16* nqwb    = (u16*)alloc(128 * 128 * 2);
  u16* nkwb    = (u16*)alloc(128 * 128 * 2);
  u16* nowTb   = (u16*)alloc(128 * 128 * 2);
  u16* rowTb   = (u16*)alloc(16 * 32 * 256 * 2);     // per-head 16 -> 32 rows

  float* xbuf  = (float*)alloc(256 * 1024 * 4);
  u16* im2cb   = (u16*)alloc(1024 * 768 * 2);
  float* hbuf  = (float*)alloc(256 * 1024 * 4);
  u16* hTb     = (u16*)alloc(1024 * 256 * 2);
  float* clsS  = (float*)alloc(20 * 1024 * 4);
  float* segS  = (float*)alloc(20 * 1024 * 4);
  float* scV   = (float*)alloc(10240 * 4);
  float* cxV   = (float*)alloc(10240 * 4);
  float* hlV   = (float*)alloc(10240 * 4);
  float* sV    = (float*)alloc(10240 * 4);
  float* eV    = (float*)alloc(10240 * 4);
  int*   tidx  = (int*)alloc(1024 * 4);
  int*   pidx  = (int*)alloc(300 * 4);
  float* bbc   = (float*)alloc(300 * 4);
  float* bbh   = (float*)alloc(300 * 4);
  int*   sidx  = (int*)alloc(300 * 4);
  int*   eidx  = (int*)alloc(300 * 4);
  u16* sppTb   = (u16*)alloc((size_t)2112 * 1024 * 2);  // 2100 -> 2112 rows
  float* Ybuf  = (float*)alloc((size_t)256 * 2100 * 4);
  u16* x2fb    = (u16*)alloc((size_t)320 * 1792 * 2);   // 300 -> 320 rows
  float* x2pre = (float*)alloc(300 * 256 * 4);
  u16* x2preb  = (u16*)alloc(320 * 256 * 2);
  u16* x2preTb = (u16*)alloc(256 * 320 * 2);            // K 300 -> 320 (zeros)
  float* qbuf  = (float*)alloc(300 * 256 * 4);
  float* kbuf  = (float*)alloc(300 * 256 * 4);
  u16* embb    = (u16*)alloc((size_t)90016 * 64 * 2);   // 90000 -> 90016 rows
  float* affw  = (float*)alloc((size_t)90000 * 16 * 4);
  u16* smb     = (u16*)alloc((size_t)16 * 320 * 320 * 2); // ld 320, K-pad zeros
  float* o_h   = (float*)alloc(300 * 256 * 4);
  u16* o_hb    = (u16*)alloc(320 * 256 * 2);
  float* relo  = (float*)alloc(300 * 256 * 4);
  float* x2r   = (float*)alloc(300 * 256 * 4);
  u16* x2rb    = (u16*)alloc(320 * 256 * 2);
  float* prob  = (float*)alloc(300 * 20 * 4);
  int*  csort  = (int*)alloc(300 * 20 * 4);
  float* ssc   = (float*)alloc(300 * 20 * 4);
  float* refined = (float*)alloc(6000 * 2 * 4);
  float* sbb   = (float*)alloc(6000 * 2 * 4);
  u16* rembb   = (u16*)alloc(320 * 256 * 2);
  float* nrank = (float*)alloc(300 * 128 * 4);
  float* roiE  = (float*)alloc(300 * 128 * 4);
  float* sroi  = (float*)alloc((size_t)6000 * 128 * 4);
  u16* sroib   = (u16*)alloc((size_t)6016 * 128 * 2);   // 6000 -> 6016 rows
  float* nemb  = (float*)alloc((size_t)6000 * 128 * 4);
  u16* nembb   = (u16*)alloc((size_t)6016 * 128 * 2);
  float* q2    = (float*)alloc((size_t)6000 * 128 * 4);
  float* k2    = (float*)alloc((size_t)6000 * 128 * 4);
  float* vp    = (float*)alloc((size_t)6000 * 128 * 4);
  float* att2  = (float*)alloc((size_t)6000 * 128 * 4);
  (void)ws_size; (void)in_sizes; (void)n_in; (void)out_size;

  hipStream_t s = stream;
  // ---- weight conversions ----
  rc3d_cvt_tr<<<CEIL(1024 * 1024, 256), 256, 0, s>>>(feature, featTb, 1024, 1024);
  rc3d_cvt<<<CEIL(256 * 1024, 256), 256, 0, s>>>(conv1_w, w1b, 256 * 1024);
  rc3d_cvt<<<CEIL(256 * 768, 256), 256, 0, s>>>(rpnc_w, wrb, 256 * 768);
  rc3d_cvt_pad<<<CEIL(32 * 256, 256), 256, 0, s>>>(rcls_w, clswb, 20, 256, 32);
  rc3d_cvt_pad<<<CEIL(32 * 256, 256), 256, 0, s>>>(rseg_w, segwb, 20, 256, 32);
  rc3d_cvt<<<CEIL(256 * 1024, 256), 256, 0, s>>>(conv2_w, w2b, 256 * 1024);
  rc3d_cvt<<<CEIL(256 * 1792, 256), 256, 0, s>>>(fc1_w, fc1wb, 256 * 1792);
  rc3d_cvt_pad<<<CEIL(32 * 256, 256), 256, 0, s>>>(cls_w, clsw21b, 21, 256, 32);
  rc3d_cvt_pad<<<CEIL(64 * 256, 256), 256, 0, s>>>(bbox_w, bboxwb, 40, 256, 64);
  rc3d_cvt<<<CEIL(128 * 256, 256), 256, 0, s>>>(fc1n_w, fc1nwb, 128 * 256);
  rc3d_cvt<<<CEIL(128 * 256, 256), 256, 0, s>>>(fc2n_w, fc2nwb, 128 * 256);
  rc3d_cvt_pad<<<CEIL(32 * 64, 256), 256, 0, s>>>(relpos_w, rposwb, 16, 64, 32);
  rc3d_cvt<<<CEIL(256 * 256, 256), 256, 0, s>>>(relq_w, rqwb, 256 * 256);
  rc3d_cvt<<<CEIL(256 * 256, 256), 256, 0, s>>>(relk_w, rkwb, 256 * 256);
  rc3d_cvt<<<CEIL(128 * 128, 256), 256, 0, s>>>(nrelq_w, nqwb, 128 * 128);
  rc3d_cvt<<<CEIL(128 * 128, 256), 256, 0, s>>>(nrelk_w, nkwb, 128 * 128);
  for (int h = 0; h < 16; ++h) {
    rc3d_cvt_tr_pad<<<CEIL(32 * 256, 256), 256, 0, s>>>(relout_w + h * 4096, rowTb + h * 8192,
                                                        256, 16, 256, 32);
    rc3d_cvt_tr<<<CEIL(128 * 8, 256), 256, 0, s>>>(nrelout_w + h * 1024, nowTb + h * 1024, 128, 8);
  }

  // ---- backbone / RPN ----
  gemm(w1b, 1024, featTb, 1024, xbuf, 1024, conv1_b, 1, 1, 256, 1024, 1024, s);
  rc3d_im2colT<<<CEIL(1024 * 768, 256), 256, 0, s>>>(xbuf, im2cb);
  gemm(wrb, 768, im2cb, 768, hbuf, 1024, rpnc_b, 1, 1, 256, 1024, 768, s);
  rc3d_cvt_tr<<<CEIL(256 * 1024, 256), 256, 0, s>>>(hbuf, hTb, 256, 1024);
  gemm(clswb, 256, hTb, 256, clsS, 1024, rcls_b, 1, 0, 20, 1024, 256, s);
  gemm(segwb, 256, hTb, 256, segS, 1024, rseg_b, 1, 0, 20, 1024, 256, s);

  // ---- proposals + NMS ----
  rc3d_scores<<<CEIL(10240, 256), 256, 0, s>>>(clsS, segS, scV, cxV, hlV, sV, eV);
  rc3d_topk<<<CEIL(10240, 256), 256, 0, s>>>(scV, tidx);
  rc3d_nms<<<1, 1024, 0, s>>>(tidx, sV, eV, cxV, hlV, pidx, bbc, bbh);
  rc3d_gather_props<<<CEIL(300, 256), 256, 0, s>>>(clsS, segS, pidx, out);
  rc3d_sidx<<<CEIL(300, 256), 256, 0, s>>>(bbc, bbh, sidx, eidx);

  // ---- ROI pool + conv2 + fc1 ----
  rc3d_roipool<<<300 * 7, 256, 0, s>>>(feature, sidx, eidx, sppTb);
  gemm(w2b, 1024, sppTb, 1024, Ybuf, 2100, conv2_b, 1, 1, 256, 2100, 1024, s);
  rc3d_x2flat<<<CEIL(300 * 1792, 256), 256, 0, s>>>(Ybuf, x2fb);
  gemm(x2fb, 1792, fc1wb, 1792, x2pre, 256, fc1_b, 2, 0, 300, 256, 1792, s);
  rc3d_cvt<<<CEIL(300 * 256, 256), 256, 0, s>>>(x2pre, x2preb, 300 * 256);
  rc3d_cvt_tr_pad<<<CEIL(256 * 320, 256), 256, 0, s>>>(x2pre, x2preTb, 300, 256, 320, 256);

  // ---- relation attention ----
  gemm(x2preb, 256, rqwb, 256, qbuf, 256, relq_b, 2, 0, 300, 256, 256, s);
  gemm(x2preb, 256, rkwb, 256, kbuf, 256, relk_b, 2, 0, 300, 256, 256, s);
  rc3d_posemb<<<CEIL(90000, 256), 256, 0, s>>>(bbc, bbh, embb);
  gemm(embb, 64, rposwb, 64, affw, 16, relpos_b, 2, 1, 90000, 16, 64, s);
  rc3d_rel_softmax<<<16 * 300, 512, 0, s>>>(affw, qbuf, kbuf, smb);
  for (int h = 0; h < 16; ++h) {
    gemm(smb + (size_t)h * 102400, 320, x2preTb, 320, o_h, 256, nullptr, 0, 0, 300, 256, 320, s);
    rc3d_cvt<<<CEIL(300 * 256, 256), 256, 0, s>>>(o_h, o_hb, 300 * 256);
    gemm(o_hb, 256, rowTb + h * 8192, 256, relo + h * 16, 256, nullptr, 0, 0, 300, 16, 256, s);
  }
  rc3d_addrelu_x2<<<CEIL(300 * 256, 256), 256, 0, s>>>(x2pre, relo, relout_b, x2r, x2rb);

  // ---- heads ----
  gemm(x2rb, 256, clsw21b, 256, out + 1200, 21, cls_b, 2, 0, 300, 21, 256, s);
  gemm(x2rb, 256, bboxwb, 256, out + 7500, 40, bbox_b, 2, 0, 300, 40, 256, s);

  // ---- learned NMS stage ----
  rc3d_prob_softmax<<<CEIL(300, 256), 256, 0, s>>>(out + 1200, prob);
  rc3d_class_sort<<<20, 512, 0, s>>>(prob, csort, ssc);
  rc3d_refined<<<CEIL(6000, 256), 256, 0, s>>>(out + 7500, bbc, bbh, refined);
  rc3d_sorted_bbox<<<CEIL(6000, 256), 256, 0, s>>>(csort, refined, sbb);
  rc3d_rank_emb<<<CEIL(300 * 256, 256), 256, 0, s>>>(rembb);
  gemm(rembb, 256, fc1nwb, 256, nrank, 128, fc1n_b, 2, 0, 300, 128, 256, s);
  gemm(x2rb, 256, fc2nwb, 256, roiE, 128, fc2n_b, 2, 0, 300, 128, 256, s);
  rc3d_sroi_nmsemb<<<CEIL(6000 * 128, 256), 256, 0, s>>>(csort, roiE, nrank, sroi, sroib, nemb, nembb);
  gemm(nembb, 128, nqwb, 128, q2, 128, nrelq_b, 2, 0, 6000, 128, 128, s);
  gemm(nembb, 128, nkwb, 128, k2, 128, nrelk_b, 2, 0, 6000, 128, 128, s);
  gemm(sroib, 128, nowTb, 128, vp, 128, nullptr, 0, 0, 6000, 128, 128, s);
  rc3d_nms_attn<<<CEIL(96000, 128), 128, 0, s>>>(sbb, q2, k2, vp, nrelpos_w, nrelpos_b, att2);
  rc3d_final<<<CEIL(6000, 256), 256, 0, s>>>(nemb, att2, nrelout_b, fc3n_w, fc3n_b, ssc, out + 19500);
}
